// ConvBlock_25108378813030
// MI455X (gfx1250) — matrix operand order repaired and checked by reading, not compiled, not run on hardware
//
#include <hip/hip_runtime.h>
#include <hip/hip_bf16.h>
#include <math.h>

// ---------------------------------------------------------------------------
// MI455X (gfx1250): conv1d(K=3,SAME) as implicit GEMM on
// V_WMMA_F32_16X16X32_BF16.  Activations for the GEMM live in a transposed,
// padded bf16 tensor Xt[b][t+1][c] so every B-fragment is a single contiguous
// 32B load per lane (no predication in the inner loop).
// ---------------------------------------------------------------------------

typedef __bf16 v16bf __attribute__((ext_vector_type(16)));
typedef float  v8f   __attribute__((ext_vector_type(8)));

#define T_LEN 512
#define TP2   (T_LEN + 2)      // +2 zero pad rows (t=-1, t=512)
#define COUT  320
#define CPAD  320              // channel-padded width of Xt
#define BATCH 256
#define NSUBJ 4
#define CT_TILES (COUT / 16)   // 20 co-tiles

// ---------------------------------------------------------------------------
// Pack conv weights w[Cout][Cin][3] (f32) into bf16 A-fragments (contiguous k order, matched by B):
//   lane<16 : M=lane,     elems 0..15 -> K=0..15
//   lane>=16: M=lane-16,  elems 0..15 -> K=16..31
// ---------------------------------------------------------------------------
__global__ __launch_bounds__(256)
void pack_weights(const float* __restrict__ w, __bf16* __restrict__ apack,
                  int cin_real, int nchunks, int total) {
  int idx = blockIdx.x * 256 + threadIdx.x;
  if (idx >= total) return;
  int e    = idx & 15;
  int lane = (idx >> 4) & 31;
  int frag = idx >> 9;
  int cc   = frag % nchunks;
  int k    = (frag / nchunks) % 3;
  int ct   = frag / (nchunks * 3);
  int m    = lane & 15;
  int kk   = e + (lane < 16 ? 0 : 16);
  int co   = ct * 16 + m;
  int ci   = cc * 32 + kk;
  float v = 0.0f;
  if (ci < cin_real) v = w[((size_t)co * cin_real + ci) * 3 + k];
  apack[idx] = (__bf16)v;
}

// ---------------------------------------------------------------------------
// Zero the t=-1 and t=512 pad rows of Xt (SAME padding).
// ---------------------------------------------------------------------------
__global__ __launch_bounds__(256)
void zero_pad_rows(__bf16* __restrict__ xt) {
  int b   = blockIdx.x >> 1;
  int row = (blockIdx.x & 1) ? (T_LEN + 1) : 0;
  __bf16* p = xt + ((size_t)b * TP2 + row) * CPAD;
  for (int c = threadIdx.x; c < CPAD; c += 256) p[c] = (__bf16)0.0f;
}

// ---------------------------------------------------------------------------
// LDS-tiled transpose: X[b][c][t] f32 -> Xt[b][t+1][c] bf16 (c zero-padded).
// ---------------------------------------------------------------------------
__global__ __launch_bounds__(256)
void transpose_in(const float* __restrict__ x, __bf16* __restrict__ xt,
                  int cin_real) {
  __shared__ float tile[64][65];
  const int b  = blockIdx.z;
  const int c0 = blockIdx.x * 64;
  const int t0 = blockIdx.y * 64;
  const int tx = threadIdx.x & 63;
  const int ty = threadIdx.x >> 6;   // 0..3
#pragma unroll
  for (int i = 0; i < 16; ++i) {
    const int cl = ty + i * 4;
    const int c  = c0 + cl;
    float v = 0.0f;
    if (c < cin_real) v = x[((size_t)b * cin_real + c) * T_LEN + t0 + tx];
    tile[cl][tx] = v;
  }
  __syncthreads();
#pragma unroll
  for (int i = 0; i < 16; ++i) {
    const int tl = ty + i * 4;
    xt[((size_t)b * TP2 + (t0 + tl + 1)) * CPAD + c0 + tx] = (__bf16)tile[tx][tl];
  }
}

// ---------------------------------------------------------------------------
// Implicit-GEMM conv1d: one wave32 per 16(co) x 64(t) tile of one batch row.
// Inner loop: 1 A-frag load + 4 B-frag loads (each one contiguous v16bf) +
// 4 WMMAs.  No branches, no selects, no padding checks.
// B-fragment layout: elem j of lane -> K = j + (lane<16?0:16), N = lane&15.
// ---------------------------------------------------------------------------
__global__ __launch_bounds__(32)
void conv1d_wmma_bf16(const __bf16* __restrict__ xt,    // [B, T+2, CPAD] bf16
                      const __bf16* __restrict__ apack, // packed weights
                      const float* __restrict__ bias,   // [COUT]
                      const float* __restrict__ resid,  // [B, COUT, T] or null
                      float* __restrict__ y,            // [B, COUT, T]
                      int nchunks) {
  const int lane = threadIdx.x;          // 0..31
  const int ct   = blockIdx.x;           // co tile
  const int t0   = blockIdx.y * 64;      // t tile base
  const int b    = blockIdx.z;           // batch row
  const int nlo  = lane & 15;
  const int hi   = lane >> 4;            // 0 or 1

  v8f acc[4] = {};

  const __bf16* xb = xt + (size_t)b * TP2 * CPAD;

  for (int k = 0; k < 3; ++k) {
    // row for nt=0: (t0 + nlo + (k-1)) + 1  (the +1 is the pad-row shift)
    const __bf16* xk = xb + (size_t)(t0 + nlo + k) * CPAD + hi * 16;
    const __bf16* ap = apack + ((size_t)(ct * 3 + k) * nchunks << 9) + (lane << 4);
    for (int cc = 0; cc < nchunks; ++cc) {
      const v16bf afrag = *(const v16bf*)(ap + ((size_t)cc << 9));
      const __bf16* xc  = xk + cc * 32;
      const v16bf bf0 = *(const v16bf*)(xc);
      const v16bf bf1 = *(const v16bf*)(xc + 16 * CPAD);
      const v16bf bf2 = *(const v16bf*)(xc + 32 * CPAD);
      const v16bf bf3 = *(const v16bf*)(xc + 48 * CPAD);
      acc[0] = __builtin_amdgcn_wmma_f32_16x16x32_bf16(false, afrag, false, bf0,
                                                       (short)0, acc[0], false, false);
      acc[1] = __builtin_amdgcn_wmma_f32_16x16x32_bf16(false, afrag, false, bf1,
                                                       (short)0, acc[1], false, false);
      acc[2] = __builtin_amdgcn_wmma_f32_16x16x32_bf16(false, afrag, false, bf2,
                                                       (short)0, acc[2], false, false);
      acc[3] = __builtin_amdgcn_wmma_f32_16x16x32_bf16(false, afrag, false, bf3,
                                                       (short)0, acc[3], false, false);
    }
  }

  // D layout: VGPR i -> M = i + (lane<16 ? 0 : 8), N = lane&15
  const int mb = ct * 16 + (hi ? 8 : 0);
#pragma unroll
  for (int nt = 0; nt < 4; ++nt) {
    const int t = t0 + nt * 16 + nlo;
#pragma unroll
    for (int i = 0; i < 8; ++i) {
      const int co = mb + i;
      const size_t idx = ((size_t)b * COUT + co) * T_LEN + t;
      float v = acc[nt][i] + bias[co];
      if (resid) v += resid[idx];
      y[idx] = v;
    }
  }
}

// ---------------------------------------------------------------------------
// Per-subject BN stats: one block per channel, deterministic reduction.
// ---------------------------------------------------------------------------
__global__ __launch_bounds__(256)
void bn_stats(const float* __restrict__ y, const int* __restrict__ subj,
              float* __restrict__ s1, float* __restrict__ s2) {
  __shared__ int   ssub[BATCH];
  __shared__ float red[256];
  const int tid = threadIdx.x;
  const int c   = blockIdx.x;
  if (tid < BATCH) ssub[tid] = subj[tid];
  __syncthreads();

  float a0 = 0.f, a1 = 0.f, a2 = 0.f, a3 = 0.f;
  float q0 = 0.f, q1 = 0.f, q2 = 0.f, q3 = 0.f;
  for (int b = 0; b < BATCH; ++b) {
    const int s = ssub[b];
    const float* p = y + ((size_t)b * COUT + c) * T_LEN;
    for (int t = tid; t < T_LEN; t += 256) {
      const float v  = p[t];
      const float vv = v * v;
      a0 += (s == 0) ? v : 0.f;  q0 += (s == 0) ? vv : 0.f;
      a1 += (s == 1) ? v : 0.f;  q1 += (s == 1) ? vv : 0.f;
      a2 += (s == 2) ? v : 0.f;  q2 += (s == 2) ? vv : 0.f;
      a3 += (s == 3) ? v : 0.f;  q3 += (s == 3) ? vv : 0.f;
    }
  }
  float vals[8] = {a0, a1, a2, a3, q0, q1, q2, q3};
#pragma unroll
  for (int r = 0; r < 8; ++r) {
    red[tid] = vals[r];
    __syncthreads();
    for (int off = 128; off > 0; off >>= 1) {
      if (tid < off) red[tid] += red[tid + off];
      __syncthreads();
    }
    if (tid == 0) {
      const int s = r & 3;
      if (r < 4) s1[s * COUT + c] = red[0];
      else       s2[s * COUT + c] = red[0];
    }
    __syncthreads();
  }
}

// ---------------------------------------------------------------------------
// scale = gamma * rsqrt(var + eps), shift = beta - mean*scale, per (s, c)
// ---------------------------------------------------------------------------
__global__ __launch_bounds__(256)
void bn_finalize(const int* __restrict__ subj, const float* __restrict__ s1,
                 const float* __restrict__ s2, const float* __restrict__ gamma,
                 const float* __restrict__ beta, float* __restrict__ scale,
                 float* __restrict__ shift) {
  __shared__ float cnt[NSUBJ];
  const int tid = threadIdx.x;
  if (tid < NSUBJ) cnt[tid] = 0.f;
  __syncthreads();
  if (tid < BATCH) atomicAdd(&cnt[subj[tid]], (float)T_LEN);  // exact in f32
  __syncthreads();
  for (int i = tid; i < NSUBJ * COUT; i += 256) {
    const int s   = i / COUT;
    const float n = fmaxf(cnt[s], 1.f);
    const float mean = s1[i] / n;
    const float var  = s2[i] / n - mean * mean;
    const float sc   = gamma[i] * rsqrtf(var + 1e-5f);
    scale[i] = sc;
    shift[i] = beta[i] - mean * sc;
  }
}

// ---------------------------------------------------------------------------
// Fused BN apply + exact-erf GELU.  Writes the f32 activation (residual /
// final output) and, if xt != null, the transposed bf16 copy for the next
// conv's B-operand (LDS-tiled so all global accesses are coalesced).
// ---------------------------------------------------------------------------
__global__ __launch_bounds__(256)
void bn_gelu_apply_t(const float* __restrict__ y, const int* __restrict__ subj,
                     const float* __restrict__ scale, const float* __restrict__ shift,
                     float* __restrict__ aout,   // f32 [B, COUT, T]
                     __bf16* __restrict__ xt) {  // bf16 [B, T+2, CPAD] or null
  __shared__ float tile[64][65];
  const int b  = blockIdx.z;
  const int c0 = blockIdx.x * 64;
  const int t0 = blockIdx.y * 64;
  const int tx = threadIdx.x & 63;
  const int ty = threadIdx.x >> 6;
  const int s  = subj[b];
#pragma unroll
  for (int i = 0; i < 16; ++i) {
    const int cl = ty + i * 4;
    const int c  = c0 + cl;
    const size_t idx = ((size_t)b * COUT + c) * T_LEN + t0 + tx;
    const float v = y[idx] * scale[s * COUT + c] + shift[s * COUT + c];
    const float g = 0.5f * v * (1.0f + erff(v * 0.70710678118654752440f));
    aout[idx] = g;
    tile[cl][tx] = g;
  }
  if (xt) {
    __syncthreads();
#pragma unroll
    for (int i = 0; i < 16; ++i) {
      const int tl = ty + i * 4;
      xt[((size_t)b * TP2 + (t0 + tl + 1)) * CPAD + c0 + tx] = (__bf16)tile[tx][tl];
    }
  }
}

// ---------------------------------------------------------------------------
// Host orchestration
// ---------------------------------------------------------------------------
extern "C" void kernel_launch(void* const* d_in, const int* in_sizes, int n_in,
                              void* d_out, int out_size, void* d_ws, size_t ws_size,
                              hipStream_t stream) {
  const float* X   = (const float*)d_in[0];   // [256, 271, 512]
  const int*   subj= (const int*)  d_in[1];   // [256]
  const float* w0  = (const float*)d_in[2];
  const float* b0  = (const float*)d_in[3];
  const float* w1  = (const float*)d_in[4];
  const float* b1  = (const float*)d_in[5];
  const float* w2  = (const float*)d_in[6];
  const float* b2  = (const float*)d_in[7];
  const float* g0  = (const float*)d_in[8];
  const float* be0 = (const float*)d_in[9];
  const float* g1  = (const float*)d_in[10];
  const float* be1 = (const float*)d_in[11];
  const float* g2  = (const float*)d_in[12];
  const float* be2 = (const float*)d_in[13];
  float* out = (float*)d_out;

  const size_t tensorBytes = (size_t)BATCH * COUT * T_LEN * sizeof(float); // 167.8 MB
  const size_t xtBytes     = (size_t)BATCH * TP2 * CPAD * sizeof(__bf16);  // 84.2 MB
  char* ws = (char*)d_ws;
  size_t off = 0;
  float*  Y   = (float*) (ws + off); off += tensorBytes;
  float*  A   = (float*) (ws + off); off += tensorBytes;
  __bf16* Xt  = (__bf16*)(ws + off); off += (xtBytes + 255) & ~(size_t)255;
  __bf16* ap0 = (__bf16*)(ws + off); off += (1u << 20);
  __bf16* ap1 = (__bf16*)(ws + off); off += (1u << 20);
  __bf16* ap2 = (__bf16*)(ws + off); off += (1u << 20);
  float* s1    = (float*)(ws + off); off += 8192;
  float* s2    = (float*)(ws + off); off += 8192;
  float* scale = (float*)(ws + off); off += 8192;
  float* shift = (float*)(ws + off); off += 8192;

  // ---- pack weights into WMMA A-fragment layout (bf16) ----
  const int nch0 = 9;   // ceil(271/32); channels 271..287 are zero-padded
  const int nch  = 10;  // 320/32
  const int tot0 = CT_TILES * 3 * nch0 * 512;
  const int tot1 = CT_TILES * 3 * nch  * 512;
  pack_weights<<<(tot0 + 255) / 256, 256, 0, stream>>>(w0, ap0, 271, nch0, tot0);
  pack_weights<<<(tot1 + 255) / 256, 256, 0, stream>>>(w1, ap1, COUT, nch, tot1);
  pack_weights<<<(tot1 + 255) / 256, 256, 0, stream>>>(w2, ap2, COUT, nch, tot1);

  // ---- build padded/transposed bf16 input ----
  zero_pad_rows<<<BATCH * 2, 256, 0, stream>>>(Xt);
  transpose_in<<<dim3(CPAD / 64, T_LEN / 64, BATCH), 256, 0, stream>>>(X, Xt, 271);

  const dim3 cgrid(CT_TILES, T_LEN / 64, BATCH);
  const dim3 agrid(COUT / 64, T_LEN / 64, BATCH);

  // ---- layer 0: conv0 (no residual) -> BN -> GELU (+transpose for layer 1) ----
  conv1d_wmma_bf16<<<cgrid, 32, 0, stream>>>(Xt, ap0, b0, nullptr, Y, nch0);
  bn_stats<<<COUT, 256, 0, stream>>>(Y, subj, s1, s2);
  bn_finalize<<<1, 256, 0, stream>>>(subj, s1, s2, g0, be0, scale, shift);
  bn_gelu_apply_t<<<agrid, 256, 0, stream>>>(Y, subj, scale, shift, A, Xt);

  // ---- layer 1: conv1 + residual -> BN -> GELU (+transpose for layer 2) ----
  conv1d_wmma_bf16<<<cgrid, 32, 0, stream>>>(Xt, ap1, b1, A, Y, nch);
  bn_stats<<<COUT, 256, 0, stream>>>(Y, subj, s1, s2);
  bn_finalize<<<1, 256, 0, stream>>>(subj, s1, s2, g1, be1, scale, shift);
  bn_gelu_apply_t<<<agrid, 256, 0, stream>>>(Y, subj, scale, shift, A, Xt);

  // ---- layer 2: conv2 + residual -> BN -> GELU -> out ----
  conv1d_wmma_bf16<<<cgrid, 32, 0, stream>>>(Xt, ap2, b2, A, Y, nch);
  bn_stats<<<COUT, 256, 0, stream>>>(Y, subj, s1, s2);
  bn_finalize<<<1, 256, 0, stream>>>(subj, s1, s2, g2, be2, scale, shift);
  bn_gelu_apply_t<<<agrid, 256, 0, stream>>>(Y, subj, scale, shift, out, nullptr);
}
